// GMN_83021717831845
// MI455X (gfx1250) — matrix operand order repaired and checked by reading, not compiled, not run on hardware
//
#include <hip/hip_runtime.h>
#include <hip/hip_bf16.h>

typedef __bf16 bf16;
typedef __attribute__((ext_vector_type(16))) __bf16 v16bf;
typedef __attribute__((ext_vector_type(8)))  float  v8f;

#define NT_TOTAL  (8 * 2 * 4096)   // 65536 nodes
#define D_FEAT    128
#define E_EDGES   524288
#define ED_FEAT   64

#define WMMA_BF16(a, b, c) \
  __builtin_amdgcn_wmma_f32_16x16x32_bf16(false, (a), false, (b), (short)0, (c), false, false)

// ---- WMMA fragment loaders (contiguous k order, same for A and B) ----

// A: 16x32 bf16 tile from LDS (row-major, stride in elements).
// lanes 0-15: rows M=lane,    K = kbase+{0..15}
// lanes16-31: rows M=lane-16, K = kbase+{16..31}
__device__ __forceinline__ v16bf load_A_lds(const bf16* base, int stride,
                                            int mbase, int kbase, int lane) {
  const int row = mbase + (lane & 15);
  const int kb  = kbase + ((lane >> 4) << 4);
  const bf16* p = base + row * stride + kb;
  v16bf a;
#pragma unroll
  for (int e = 0; e < 8; ++e) a[e] = p[e];
#pragma unroll
  for (int e = 0; e < 8; ++e) a[8 + e] = p[8 + e];
  return a;
}

// B: packed fragment-major weights. For tile (nt, kt) the 32x16 B tile is
// stored lane-major: 32 lanes x 16 contiguous bf16 (32B per lane, coalesced).
// element e of lane = W[(kt*32 + (lane>>4)*16 + e)][nt*16 + (lane&15)]
__device__ __forceinline__ v16bf load_B_pk(const bf16* P, int nkt, int kt,
                                           int nt, int lane) {
  const bf16* p = P + ((((size_t)nt * nkt + kt) << 5) + lane) * 16;
  return *(const v16bf*)p;
}

// 4 M-tiles (64 rows) x one 16-col N-tile, accumulate over K = nkt*32.
template <int NKT>
__device__ __forceinline__ void mm64x16(const bf16* A, int lda, const bf16* Wp,
                                        const float* bias, int nt, int lane,
                                        v8f acc[4]) {
  const float bv = bias[(nt << 4) + (lane & 15)];
#pragma unroll
  for (int mt = 0; mt < 4; ++mt)
#pragma unroll
    for (int v = 0; v < 8; ++v) acc[mt][v] = bv;
  for (int kt = 0; kt < NKT; ++kt) {
    v16bf b = load_B_pk(Wp, NKT, kt, nt, lane);
#pragma unroll
    for (int mt = 0; mt < 4; ++mt) {
      v16bf a = load_A_lds(A, lda, mt * 16, kt * 32, lane);
      acc[mt] = WMMA_BF16(a, b, acc[mt]);
    }
  }
}

// C/D 16x16 f32: element v -> row mbase + v + 8*hi, col = nt*16 + (lane&15)
__device__ __forceinline__ void store_relu_lds(v8f acc[4], bf16* dst,
                                               int col, int hi) {
#pragma unroll
  for (int mt = 0; mt < 4; ++mt) {
    const int rbase = mt * 16 + (hi << 3);
#pragma unroll
    for (int v = 0; v < 8; ++v) {
      float x = acc[mt][v];
      dst[(rbase + v) * 128 + col] = (bf16)(x > 0.f ? x : 0.f);
    }
  }
}

// ---------------- prep kernels ----------------

// Pack W f32 [Ktot][128] row-major -> bf16 fragment-major (see load_B_pk).
__global__ __launch_bounds__(256) void pack_w_kernel(const float* __restrict__ W,
                                                     bf16* __restrict__ P,
                                                     int Ktot) {
  const int idx = blockIdx.x * 256 + threadIdx.x;
  if (idx >= Ktot * 128) return;
  const int e    = idx & 15;
  const int lane = (idx >> 4) & 31;
  const int t    = idx >> 9;              // nt*nkt + kt
  const int nkt  = Ktot >> 5;
  const int nt   = t / nkt;
  const int kt   = t - nt * nkt;
  const int col  = (nt << 4) + (lane & 15);
  const int k    = (kt << 5) + ((lane >> 4) << 4) + e;
  P[idx] = (bf16)W[k * 128 + col];
}

__global__ __launch_bounds__(256) void init_ns_kernel(const float* __restrict__ nf,
                                                      float* __restrict__ ns,
                                                      bf16* __restrict__ nsb) {
  size_t i = (size_t)blockIdx.x * 256 + threadIdx.x;
  float v = nf[i];
  ns[i]  = v;
  nsb[i] = (bf16)v;
}

// convert 8 consecutive f32 -> 8 bf16 and store as one 16B chunk
__device__ __forceinline__ void cvt8_store(const float* s0, bf16* d) {
  bf16 tmp[8];
#pragma unroll
  for (int k = 0; k < 8; ++k) tmp[k] = (bf16)s0[k];
  *(int4*)d = *(const int4*)tmp;
}

// ---------------- message MLP + scatter ----------------
// block = 256 threads (8 waves), 64 edges. LDS: in[64][320] bf16, h1[64][128] bf16,
// src/dst idx[64]. h2 reuses the in-region.
__global__ __launch_bounds__(256) void msg_kernel(
    const bf16* __restrict__ ns_bf, const int* __restrict__ ev,
    const float* __restrict__ ef,
    const bf16* __restrict__ w1, const bf16* __restrict__ w2,
    const bf16* __restrict__ w3,
    const float* __restrict__ b1, const float* __restrict__ b2,
    const float* __restrict__ b3,
    float* __restrict__ summed) {
  extern __shared__ char smem[];
  bf16* s_in = (bf16*)smem;             // 64*320
  bf16* s_h1 = s_in + 64 * 320;         // 64*128
  bf16* s_h2 = s_in;                    // reuse
  int*  s_si = (int*)(s_h1 + 64 * 128); // 64
  int*  s_di = s_si + 64;               // 64

  const int e0  = blockIdx.x * 64;
  const int tid = threadIdx.x;

  // preload edge endpoint indices (coalesced b64 loads by first 64 threads)
  if (tid < 64) {
    s_si[tid] = ev[2 * (e0 + tid) + 0];
    s_di[tid] = ev[2 * (e0 + tid) + 1];
  }
  __syncthreads();

  { // stage, fully 16B-chunked:
    // src rows: 64 rows x 16 chunks (128 bf16 = 256B)
    for (int idx = tid; idx < 64 * 16; idx += 256) {
      const int row = idx >> 4, j = idx & 15;
      ((int4*)(s_in + row * 320))[j] =
          ((const int4*)(ns_bf + (size_t)s_si[row] * 128))[j];
    }
    // dst rows
    for (int idx = tid; idx < 64 * 16; idx += 256) {
      const int row = idx >> 4, j = idx & 15;
      ((int4*)(s_in + row * 320 + 128))[j] =
          ((const int4*)(ns_bf + (size_t)s_di[row] * 128))[j];
    }
    // edge features: 64 rows x 8 chunks of 8 cols (f32 -> bf16)
    for (int idx = tid; idx < 64 * 8; idx += 256) {
      const int row = idx >> 3, ch = idx & 7;
      cvt8_store(ef + (size_t)(e0 + row) * 64 + ch * 8,
                 s_in + row * 320 + 256 + ch * 8);
    }
  }
  __syncthreads();

  const int lane = tid & 31;
  const int nt   = tid >> 5;
  const int colL = lane & 15;
  const int hi   = lane >> 4;
  const int col  = (nt << 4) + colL;

  v8f acc[4];
  // layer 1: 320 -> 128, relu
  mm64x16<10>(s_in, 320, w1, b1, nt, lane, acc);
  store_relu_lds(acc, s_h1, col, hi);
  __syncthreads();
  // layer 2: 128 -> 128, relu (writes into the staged-input region)
  mm64x16<4>(s_h1, 128, w2, b2, nt, lane, acc);
  store_relu_lds(acc, s_h2, col, hi);
  __syncthreads();
  // layer 3: 128 -> 128, then segment-sum scatter to both endpoints
  mm64x16<4>(s_h2, 128, w3, b3, nt, lane, acc);
#pragma unroll
  for (int mt = 0; mt < 4; ++mt) {
    const int rbase = mt * 16 + (hi << 3);
#pragma unroll
    for (int v = 0; v < 8; ++v) {
      const int   row = rbase + v;
      const float val = acc[mt][v];
      atomicAdd(&summed[(size_t)s_si[row] * 128 + col], val);
      atomicAdd(&summed[(size_t)s_di[row] * 128 + col], val);
    }
  }
}

// ---------------- update MLP ----------------
// block = 64 nodes. upd_in = [ns | summed | ns - ns(partner)] (384 wide).
__global__ __launch_bounds__(256) void upd_kernel(
    const float* __restrict__ ns_f, const float* __restrict__ summed,
    const bf16* __restrict__ w1, const bf16* __restrict__ w2,
    const bf16* __restrict__ w3,
    const float* __restrict__ b1, const float* __restrict__ b2,
    const float* __restrict__ b3,
    float* __restrict__ ns_out, bf16* __restrict__ nsbf_out) {
  extern __shared__ char smem[];
  bf16* s_in = (bf16*)smem;       // 64*384
  bf16* s_h1 = s_in + 64 * 384;   // 64*128
  bf16* s_h2 = s_in;              // reuse

  const int i0  = blockIdx.x * 64;
  const int tid = threadIdx.x;

  { // stage: 64 rows x 48 chunks of 8 cols; chunk regions are uniform
    for (int idx = tid; idx < 64 * 48; idx += 256) {
      const int row = idx / 48, ch = idx - row * 48;
      const int i = i0 + row;
      bf16* d = s_in + row * 384 + ch * 8;
      if (ch < 16) {                       // ns columns
        cvt8_store(ns_f + (size_t)i * 128 + ch * 8, d);
      } else if (ch < 32) {                // summed columns
        cvt8_store(summed + (size_t)i * 128 + (ch - 16) * 8, d);
      } else {                             // att = ns - ns(partner)
        const int bb = i >> 13, rr = (i >> 12) & 1, nn = i & 4095;
        const int p  = (bb << 13) | ((1 ^ rr) << 12) | nn;
        const float* a  = ns_f + (size_t)i * 128 + (ch - 32) * 8;
        const float* pp = ns_f + (size_t)p * 128 + (ch - 32) * 8;
        bf16 tmp[8];
#pragma unroll
        for (int k = 0; k < 8; ++k) tmp[k] = (bf16)(a[k] - pp[k]);
        *(int4*)d = *(const int4*)tmp;
      }
    }
  }
  __syncthreads();

  const int lane = tid & 31;
  const int nt   = tid >> 5;
  const int colL = lane & 15;
  const int hi   = lane >> 4;
  const int col  = (nt << 4) + colL;

  v8f acc[4];
  mm64x16<12>(s_in, 384, w1, b1, nt, lane, acc);   // 384 -> 128, relu
  store_relu_lds(acc, s_h1, col, hi);
  __syncthreads();
  mm64x16<4>(s_h1, 128, w2, b2, nt, lane, acc);    // 128 -> 128, relu
  store_relu_lds(acc, s_h2, col, hi);
  __syncthreads();
  mm64x16<4>(s_h2, 128, w3, b3, nt, lane, acc);    // 128 -> 128
#pragma unroll
  for (int mt = 0; mt < 4; ++mt) {
    const int rbase = mt * 16 + (hi << 3);
#pragma unroll
    for (int v = 0; v < 8; ++v) {
      const size_t gi = (size_t)(i0 + rbase + v) * 128 + col;
      const float val = acc[mt][v];
      ns_out[gi]   = val;
      nsbf_out[gi] = (bf16)val;
    }
  }
}

// ---------------- gate + per-graph column sums ----------------
__global__ __launch_bounds__(256) void gate_kernel(
    const bf16* __restrict__ ns_bf, const float* __restrict__ ns_f,
    const bf16* __restrict__ gw, const float* __restrict__ gb,
    float* __restrict__ gs) {
  extern __shared__ char smem[];
  bf16* s_in = (bf16*)smem;  // 64*128
  const int i0  = blockIdx.x * 64;
  const int tid = threadIdx.x;
  for (int idx = tid; idx < 64 * 16; idx += 256)  // 16B chunks
    ((int4*)s_in)[idx] = ((const int4*)(ns_bf + (size_t)i0 * 128))[idx];
  __syncthreads();

  const int lane = tid & 31;
  const int nt   = tid >> 5;
  const int colL = lane & 15;
  const int hi   = lane >> 4;
  const int col  = (nt << 4) + colL;

  v8f acc[4];
  mm64x16<4>(s_in, 128, gw, gb, nt, lane, acc);
  float colacc = 0.f;
#pragma unroll
  for (int mt = 0; mt < 4; ++mt) {
    const int rbase = mt * 16 + (hi << 3);
#pragma unroll
    for (int v = 0; v < 8; ++v) {
      const float sig = 1.f / (1.f + __expf(-acc[mt][v]));
      colacc += ns_f[(size_t)(i0 + rbase + v) * 128 + col] * sig;
    }
  }
  // all 64 rows of this block belong to one (batch, side) segment
  atomicAdd(&gs[(size_t)(i0 >> 12) * 128 + col], colacc);
}

// ---------------- tiny final MLP: [16,128] -> relu[16,256] -> [16,128] ----------------
__global__ __launch_bounds__(256) void final_kernel(
    const float* __restrict__ gs,
    const float* __restrict__ w1, const float* __restrict__ b1,
    const float* __restrict__ w2, const float* __restrict__ b2,
    float* __restrict__ out) {
  __shared__ float s_h[16 * 256];
  const int tid = threadIdx.x;
  for (int idx = tid; idx < 16 * 256; idx += 256) {
    const int i = idx >> 8, j = idx & 255;
    float acc = b1[j];
    for (int k = 0; k < 128; ++k) acc += gs[i * 128 + k] * w1[k * 256 + j];
    s_h[idx] = acc > 0.f ? acc : 0.f;
  }
  __syncthreads();
  for (int idx = tid; idx < 16 * 128; idx += 256) {
    const int i = idx >> 7, j = idx & 127;
    float acc = b2[j];
    for (int k = 0; k < 256; ++k) acc += s_h[i * 256 + k] * w2[k * 128 + j];
    out[idx] = acc;
  }
}

// ---------------- host orchestration ----------------

extern "C" void kernel_launch(void* const* d_in, const int* in_sizes, int n_in,
                              void* d_out, int out_size, void* d_ws, size_t ws_size,
                              hipStream_t stream) {
  const float* nf  = (const float*)d_in[0];
  const float* ef  = (const float*)d_in[1];
  const int*   ev  = (const int*)d_in[2];
  const float* mw1 = (const float*)d_in[3];
  const float* mb1 = (const float*)d_in[4];
  const float* mw2 = (const float*)d_in[5];
  const float* mb2 = (const float*)d_in[6];
  const float* mw3 = (const float*)d_in[7];
  const float* mb3 = (const float*)d_in[8];
  const float* uw1 = (const float*)d_in[9];
  const float* ub1 = (const float*)d_in[10];
  const float* uw2 = (const float*)d_in[11];
  const float* ub2 = (const float*)d_in[12];
  const float* uw3 = (const float*)d_in[13];
  const float* ub3 = (const float*)d_in[14];
  const float* gw  = (const float*)d_in[15];
  const float* gb  = (const float*)d_in[16];
  const float* aw1 = (const float*)d_in[17];
  const float* ab1 = (const float*)d_in[18];
  const float* aw2 = (const float*)d_in[19];
  const float* ab2 = (const float*)d_in[20];

  const size_t ND = (size_t)NT_TOTAL * D_FEAT;
  char* ws = (char*)d_ws;
  float* ns_a   = (float*)ws;
  float* ns_b   = ns_a + ND;
  bf16*  ns_bf  = (bf16*)(ns_b + ND);
  float* summed = (float*)(ns_bf + ND);
  bf16*  wbf    = (bf16*)(summed + ND);
  bf16* w1m = wbf;                     // 320*128 packed
  bf16* w2m = w1m + 320 * 128;         // 128*128 packed
  bf16* w3m = w2m + 128 * 128;
  bf16* u1m = w3m + 128 * 128;         // 384*128 packed
  bf16* u2m = u1m + 384 * 128;
  bf16* u3m = u2m + 128 * 128;
  bf16* gwm = u3m + 128 * 128;
  float* gs = (float*)(gwm + 128 * 128);

  auto pack = [&](const float* s, bf16* d, int ktot) {
    pack_w_kernel<<<(ktot * 128 + 255) / 256, 256, 0, stream>>>(s, d, ktot);
  };
  pack(mw1, w1m, 320);
  pack(mw2, w2m, 128);
  pack(mw3, w3m, 128);
  pack(uw1, u1m, 384);
  pack(uw2, u2m, 128);
  pack(uw3, u3m, 128);
  pack(gw,  gwm, 128);

  init_ns_kernel<<<(int)(ND / 256), 256, 0, stream>>>(nf, ns_a, ns_bf);

  const size_t msg_smem = 64 * 320 * sizeof(bf16) + 64 * 128 * sizeof(bf16) + 128 * sizeof(int);
  const size_t upd_smem = 64 * 384 * sizeof(bf16) + 64 * 128 * sizeof(bf16);

  float* cur = ns_a;
  float* nxt = ns_b;
  for (int layer = 0; layer < 3; ++layer) {
    hipMemsetAsync(summed, 0, ND * sizeof(float), stream);
    msg_kernel<<<E_EDGES / 64, 256, msg_smem, stream>>>(
        ns_bf, ev, ef, w1m, w2m, w3m, mb1, mb2, mb3, summed);
    upd_kernel<<<NT_TOTAL / 64, 256, upd_smem, stream>>>(
        cur, summed, u1m, u2m, u3m, ub1, ub2, ub3, nxt, ns_bf);
    float* t = cur; cur = nxt; nxt = t;
  }

  hipMemsetAsync(gs, 0, 16 * 128 * sizeof(float), stream);
  gate_kernel<<<NT_TOTAL / 64, 256, 64 * 128 * sizeof(bf16), stream>>>(
      ns_bf, cur, gwm, gb, gs);
  final_kernel<<<1, 256, 0, stream>>>(gs, aw1, ab1, aw2, ab2, (float*)d_out);
}
